// TriangleAttention_481036337744
// MI455X (gfx1250) — matrix operand order repaired and checked by reading, not compiled, not run on hardware
//
#include <hip/hip_runtime.h>
#include <hip/hip_bf16.h>

typedef __attribute__((ext_vector_type(16))) _Float16 v16h;
typedef __attribute__((ext_vector_type(8)))  float    v8f;

#define LSEQ   256
#define DMODEL 128
#define NHEAD  4
#define DHEAD  32
#define STR    136   // LDS row stride (halves) for 256x128 activation buffers
#define PSTR   40    // per-wave P-scratch row stride (halves)
#define SMEM_HALVES (4 * LSEQ * STR + 8 * 16 * PSTR)

static __device__ __forceinline__ v8f wmma_f16(v16h a, v16h b, v8f c) {
  // D = A(16x32 f16) * B(32x16 f16) + C(16x16 f32)
  return __builtin_amdgcn_wmma_f32_16x16x32_f16(false, a, false, b, (short)0, c,
                                                false, false);
}

// A fragment (16x32 f16): A[m][k], m = lane%16, element t -> k = t + 16*(lane/16)
static __device__ __forceinline__ v16h load_afrag(const _Float16* base, int row0,
                                                  int col0, int stride, int ln16, int g) {
  const _Float16* r = base + (row0 + ln16) * stride + col0 + g * 16;
  v16h a;
#pragma unroll
  for (int t = 0; t < 16; ++t) a[t] = r[t];
  return a;
}

// B fragment (32x16 f16): B[k][n], n = lane%16, k = t + 16*(lane/16)
// transposed source: B[k][n] = buf[(rbase+n)*stride + cbase + k]   (e.g. K^T)
static __device__ __forceinline__ v16h load_bfrag_T(const _Float16* buf, int rbase,
                                                    int cbase, int stride, int ln16, int g) {
  const _Float16* p = buf + (rbase + ln16) * stride + cbase + g * 16;
  v16h b;
#pragma unroll
  for (int t = 0; t < 16; ++t) b[t] = p[t];
  return b;
}

// natural source: B[k][n] = buf[(kbase+k)*stride + cbase + n]      (e.g. V)
static __device__ __forceinline__ v16h load_bfrag_N(const _Float16* buf, int kbase,
                                                    int cbase, int stride, int ln16, int g) {
  const _Float16* p = buf + (kbase + g * 16) * stride + cbase + ln16;
  v16h b;
#pragma unroll
  for (int t = 0; t < 16; ++t) b[t] = p[t * stride];
  return b;
}

// B fragment from fp32 row-major weight W[e][d] (128x128): B[k][n] = W[(eb*16+n)*128 + k0+k]
static __device__ __forceinline__ v16h load_bfrag_W(const float* W, int eb, int k0,
                                                    int ln16, int g) {
  const float* p = W + (eb * 16 + ln16) * DMODEL + k0 + g * 16;
  v16h b;
#pragma unroll
  for (int t = 0; t < 16; ++t) b[t] = (_Float16)p[t];
  return b;
}

__global__ __launch_bounds__(256)
void tri_attn(const float* __restrict__ pair, const unsigned char* __restrict__ pmask,
              const float* __restrict__ ln_g, const float* __restrict__ ln_b,
              const float* __restrict__ Wq, const float* __restrict__ bq,
              const float* __restrict__ Wk, const float* __restrict__ bk,
              const float* __restrict__ Wv, const float* __restrict__ bv,
              const float* __restrict__ Wo, const float* __restrict__ bo,
              const float* __restrict__ Wg, const float* __restrict__ bg,
              float* __restrict__ out) {
  extern __shared__ _Float16 sm[];
  _Float16* PN = sm;                  // pn (LN output), later overwritten by gate
  _Float16* QO = PN + LSEQ * STR;     // Q, later overwritten per (jb, head) by O
  _Float16* KB = QO + LSEQ * STR;     // K
  _Float16* VB = KB + LSEQ * STR;     // V
  _Float16* PS = VB + LSEQ * STR;     // per-wave 16x32 P tiles

  const int i    = blockIdx.x;
  const int tid  = threadIdx.x;
  const int lane = tid & 31;
  const int wave = tid >> 5;
  const int g    = lane >> 4;         // half-wave group
  const int ln16 = lane & 15;

  // Warm L2 with the weight matrices (global_prefetch_b8).
  __builtin_prefetch(Wq + tid * 64, 0, 3);
  __builtin_prefetch(Wk + tid * 64, 0, 3);
  __builtin_prefetch(Wv + tid * 64, 0, 3);
  __builtin_prefetch(Wo + tid * 64, 0, 3);
  __builtin_prefetch(Wg + tid * 64, 0, 3);

  const float* X = pair + (size_t)i * LSEQ * DMODEL;

  // ---- LayerNorm: each wave owns rows [wave*32, wave*32+32); coalesced float4 ----
  for (int r = wave * 32; r < wave * 32 + 32; ++r) {
    float4 v = ((const float4*)(X + r * DMODEL))[lane];
    float s  = v.x + v.y + v.z + v.w;
    float s2 = v.x * v.x + v.y * v.y + v.z * v.z + v.w * v.w;
#pragma unroll
    for (int off = 16; off; off >>= 1) {
      s  += __shfl_xor(s,  off, 32);
      s2 += __shfl_xor(s2, off, 32);
    }
    float mu   = s * (1.0f / DMODEL);
    float var  = s2 * (1.0f / DMODEL) - mu * mu;
    float rstd = rsqrtf(var + 1e-5f);
    float4 gm = ((const float4*)ln_g)[lane];
    float4 bt = ((const float4*)ln_b)[lane];
    _Float16* dst = PN + r * STR + lane * 4;
    dst[0] = (_Float16)((v.x - mu) * rstd * gm.x + bt.x);
    dst[1] = (_Float16)((v.y - mu) * rstd * gm.y + bt.y);
    dst[2] = (_Float16)((v.z - mu) * rstd * gm.z + bt.z);
    dst[3] = (_Float16)((v.w - mu) * rstd * gm.w + bt.w);
  }
  // All A-fragment reads below touch only this wave's own rows -> no barrier yet.

  const int jb0 = wave * 2;

  // ---- Q / K / V projections: C = pn @ W^T + b  (1/sqrt(DH) folded into Q) ----
  for (int m = 0; m < 3; ++m) {
    const float* W    = (m == 0) ? Wq : (m == 1) ? Wk : Wv;
    const float* bias = (m == 0) ? bq : (m == 1) ? bk : bv;
    _Float16*    Dst  = (m == 0) ? QO : (m == 1) ? KB : VB;
    const float  scl  = (m == 0) ? 0.17677669529663688f : 1.0f;
    for (int jb = jb0; jb < jb0 + 2; ++jb) {
      v16h a0 = load_afrag(PN, jb * 16, 0,  STR, ln16, g);
      v16h a1 = load_afrag(PN, jb * 16, 32, STR, ln16, g);
      v16h a2 = load_afrag(PN, jb * 16, 64, STR, ln16, g);
      v16h a3 = load_afrag(PN, jb * 16, 96, STR, ln16, g);
#pragma unroll
      for (int eb = 0; eb < 8; ++eb) {
        v8f c = {};
        c = wmma_f16(a0, load_bfrag_W(W, eb, 0,  ln16, g), c);
        c = wmma_f16(a1, load_bfrag_W(W, eb, 32, ln16, g), c);
        c = wmma_f16(a2, load_bfrag_W(W, eb, 64, ln16, g), c);
        c = wmma_f16(a3, load_bfrag_W(W, eb, 96, ln16, g), c);
        float bb = bias[eb * 16 + ln16];
#pragma unroll
        for (int r = 0; r < 8; ++r) {
          int row = jb * 16 + r + 8 * g;
          Dst[row * STR + eb * 16 + ln16] = (_Float16)((c[r] + bb) * scl);
        }
      }
    }
  }

  // ---- gate = sigmoid(pn @ Wg^T + bg); overwrite wave-local PN rows in place ----
  for (int jb = jb0; jb < jb0 + 2; ++jb) {
    v16h a0 = load_afrag(PN, jb * 16, 0,  STR, ln16, g);
    v16h a1 = load_afrag(PN, jb * 16, 32, STR, ln16, g);
    v16h a2 = load_afrag(PN, jb * 16, 64, STR, ln16, g);
    v16h a3 = load_afrag(PN, jb * 16, 96, STR, ln16, g);
    v8f acc[8];
#pragma unroll
    for (int eb = 0; eb < 8; ++eb) {
      v8f c = {};
      c = wmma_f16(a0, load_bfrag_W(Wg, eb, 0,  ln16, g), c);
      c = wmma_f16(a1, load_bfrag_W(Wg, eb, 32, ln16, g), c);
      c = wmma_f16(a2, load_bfrag_W(Wg, eb, 64, ln16, g), c);
      c = wmma_f16(a3, load_bfrag_W(Wg, eb, 96, ln16, g), c);
      acc[eb] = c;
    }
#pragma unroll
    for (int eb = 0; eb < 8; ++eb) {
      float bb = bg[eb * 16 + ln16];
#pragma unroll
      for (int r = 0; r < 8; ++r) {
        int row = jb * 16 + r + 8 * g;
        float z = acc[eb][r] + bb;
        PN[row * STR + eb * 16 + ln16] = (_Float16)(1.0f / (1.0f + __expf(-z)));
      }
    }
  }

  __syncthreads();  // make all K / V rows visible to every wave

  // ---- flash attention per (jb, head): scores 16x256, online softmax, O = P@V ----
  _Float16* myPS = PS + wave * 16 * PSTR;
  for (int jb = jb0; jb < jb0 + 2; ++jb) {
    for (int h = 0; h < NHEAD; ++h) {
      const int c0 = h * DHEAD;
      v16h aq = load_afrag(QO, jb * 16, c0, STR, ln16, g);  // 16 rows x K=32 (= DH)
      v8f o0 = {}, o1 = {};
      float Mx[8], Sm[8];
#pragma unroll
      for (int r = 0; r < 8; ++r) { Mx[r] = -3.0e38f; Sm[r] = 0.0f; }
#pragma unroll 1
      for (int c = 0; c < 8; ++c) {  // 32 key columns per step
        v8f z0 = {}, z1 = {};
        v8f s0 = wmma_f16(aq, load_bfrag_T(KB, c * 32,      c0, STR, ln16, g), z0);
        v8f s1 = wmma_f16(aq, load_bfrag_T(KB, c * 32 + 16, c0, STR, ln16, g), z1);
#pragma unroll
        for (int r = 0; r < 8; ++r) {
          int row = jb * 16 + r + 8 * g;
          int kc  = c * 32 + ln16;
          float v0 = pmask[row * LSEQ + kc]      ? s0[r] : -1.0e9f;
          float v1 = pmask[row * LSEQ + kc + 16] ? s1[r] : -1.0e9f;
          float mx = fmaxf(v0, v1);
#pragma unroll
          for (int off = 8; off; off >>= 1) mx = fmaxf(mx, __shfl_xor(mx, off, 16));
          float Mn = fmaxf(Mx[r], mx);
          float p0 = __expf(v0 - Mn);
          float p1 = __expf(v1 - Mn);
          float rs = p0 + p1;
#pragma unroll
          for (int off = 8; off; off >>= 1) rs += __shfl_xor(rs, off, 16);
          float corr = __expf(Mx[r] - Mn);
          Sm[r] = Sm[r] * corr + rs;
          Mx[r] = Mn;
          o0[r] *= corr;
          o1[r] *= corr;
          _Float16* ps = myPS + (r + 8 * g) * PSTR;  // D-layout -> row-major 16x32
          ps[ln16]      = (_Float16)p0;
          ps[ln16 + 16] = (_Float16)p1;
        }
        asm volatile("s_wait_dscnt 0" ::: "memory");   // wave-private LDS RAW guard
        v16h ap = load_afrag(myPS, 0, 0, PSTR, ln16, g);
        o0 = wmma_f16(ap, load_bfrag_N(VB, c * 32, c0,      STR, ln16, g), o0);
        o1 = wmma_f16(ap, load_bfrag_N(VB, c * 32, c0 + 16, STR, ln16, g), o1);
      }
#pragma unroll
      for (int r = 0; r < 8; ++r) {  // normalize, store O over consumed Q columns
        int row = jb * 16 + r + 8 * g;
        float inv = 1.0f / Sm[r];
        QO[row * STR + c0 + ln16]      = (_Float16)(o0[r] * inv);
        QO[row * STR + c0 + 16 + ln16] = (_Float16)(o1[r] * inv);
      }
    }
  }
  // O and gate are wave-local -> no barrier needed before the final projection.

  // ---- out = (O @ Wo^T + bo) * gate ----
  for (int jb = jb0; jb < jb0 + 2; ++jb) {
    v16h a0 = load_afrag(QO, jb * 16, 0,  STR, ln16, g);
    v16h a1 = load_afrag(QO, jb * 16, 32, STR, ln16, g);
    v16h a2 = load_afrag(QO, jb * 16, 64, STR, ln16, g);
    v16h a3 = load_afrag(QO, jb * 16, 96, STR, ln16, g);
#pragma unroll
    for (int eb = 0; eb < 8; ++eb) {
      v8f c = {};
      c = wmma_f16(a0, load_bfrag_W(Wo, eb, 0,  ln16, g), c);
      c = wmma_f16(a1, load_bfrag_W(Wo, eb, 32, ln16, g), c);
      c = wmma_f16(a2, load_bfrag_W(Wo, eb, 64, ln16, g), c);
      c = wmma_f16(a3, load_bfrag_W(Wo, eb, 96, ln16, g), c);
      float bb = bo[eb * 16 + ln16];
#pragma unroll
      for (int r = 0; r < 8; ++r) {
        int row = jb * 16 + r + 8 * g;
        float gt = (float)PN[row * STR + eb * 16 + ln16];
        out[((size_t)i * LSEQ + row) * DMODEL + eb * 16 + ln16] = (c[r] + bb) * gt;
      }
    }
  }
}

extern "C" void kernel_launch(void* const* d_in, const int* in_sizes, int n_in,
                              void* d_out, int out_size, void* d_ws, size_t ws_size,
                              hipStream_t stream) {
  (void)in_sizes; (void)n_in; (void)out_size; (void)d_ws; (void)ws_size;
  const float*         pair  = (const float*)d_in[0];
  const unsigned char* pmask = (const unsigned char*)d_in[1];  // jnp bool = 1 byte
  const float* lng = (const float*)d_in[2];
  const float* lnb = (const float*)d_in[3];
  const float* Wq  = (const float*)d_in[4];
  const float* bq  = (const float*)d_in[5];
  const float* Wk  = (const float*)d_in[6];
  const float* bk  = (const float*)d_in[7];
  const float* Wv  = (const float*)d_in[8];
  const float* bv  = (const float*)d_in[9];
  const float* Wo  = (const float*)d_in[10];
  const float* bo  = (const float*)d_in[11];
  const float* Wg  = (const float*)d_in[12];
  const float* bg  = (const float*)d_in[13];
  float* out = (float*)d_out;

  size_t smem = (size_t)SMEM_HALVES * sizeof(_Float16);  // 288,768 B (< 320 KB WGP LDS)
  hipFuncSetAttribute((const void*)tri_attn,
                      hipFuncAttributeMaxDynamicSharedMemorySize, (int)smem);
  tri_attn<<<LSEQ, 256, smem, stream>>>(pair, pmask, lng, lnb, Wq, bq, Wk, bk,
                                        Wv, bv, Wo, bo, Wg, bg, out);
}
